// BayesianRNN_37383395345224
// MI455X (gfx1250) — matrix operand order repaired and checked by reading, not compiled, not run on hardware
//
#include <hip/hip_runtime.h>
#include <hip/hip_bf16.h>

// ---------------------------------------------------------------- constants
#define T_STEPS 384
#define BATCH   128
#define IN_DIM  128
#define HID     1024
#define GATES   4096   // 4*HID

typedef __attribute__((ext_vector_type(16))) __bf16 v16bf;
typedef __attribute__((ext_vector_type(8)))  float  v8f;
typedef __attribute__((ext_vector_type(8)))  unsigned short u16x8;

struct BF16Frag { u16x8 lo, hi; };

__device__ __forceinline__ v16bf load_frag16(const unsigned short* p0,
                                             const unsigned short* p1) {
    BF16Frag t;
    t.lo = *(const u16x8*)p0;
    t.hi = *(const u16x8*)p1;
    return __builtin_bit_cast(v16bf, t);
}

// ------- CDNA5 async global->LDS copy (ASYNCcnt path), 16B per lane --------
// IOFFSET applies to BOTH the LDS and global address (ISA 08_async_tensor §4.4),
// so one base pair + offset:16 moves 32 contiguous bytes per lane.
__device__ __forceinline__ void async_copy_b128(unsigned lds_off, const void* gptr) {
    asm volatile("global_load_async_to_lds_b128 %0, %1, off"
                 :: "v"(lds_off), "v"(gptr) : "memory");
}
__device__ __forceinline__ void async_copy_b128_off16(unsigned lds_off, const void* gptr) {
    asm volatile("global_load_async_to_lds_b128 %0, %1, off offset:16"
                 :: "v"(lds_off), "v"(gptr) : "memory");
}
__device__ __forceinline__ void s_wait_asynccnt_0() {
    asm volatile("s_wait_asynccnt 0x0" ::: "memory");
}
__device__ __forceinline__ void s_wait_asynccnt_4() {
    asm volatile("s_wait_asynccnt 0x4" ::: "memory");
}
// low 32 bits of a generic pointer to __shared__ == LDS byte offset
template <typename T>
__device__ __forceinline__ unsigned lds_off32(T* p) {
    return (unsigned)(size_t)(const void*)p;
}

__device__ __forceinline__ float sigm(float x) {
    return 1.0f / (1.0f + __expf(-x));
}
__device__ __forceinline__ float tanh_fast(float x) {
    float xx = fminf(fmaxf(x, -15.0f), 15.0f);
    float e = __expf(2.0f * xx);
    return (e - 1.0f) / (e + 1.0f);
}

// ---------------------------------------------------------------- init
__global__ void init_kernel(unsigned short* h0, unsigned short* h1,
                            float* cbuf, float* acc, int n) {
    int i = blockIdx.x * blockDim.x + threadIdx.x;
    if (i < n) { h0[i] = 0; h1[i] = 0; cbuf[i] = 0.0f; }
    if (i < 8) acc[i] = 0.0f;
}

// ---------------------------------------------------------------- weight sampling + KL partials
// acc[0] += sum(-log(sigma) - 0.5*eps^2), acc[1] += sum(w^2); also writes bf16 w.
__global__ __launch_bounds__(256)
void sample_matrix_kernel(const float* __restrict__ mu, const float* __restrict__ rho,
                          const float* __restrict__ eps, size_t n,
                          unsigned short* __restrict__ wbf, float* __restrict__ acc) {
    __bf16* wb = (__bf16*)wbf;
    float lpost = 0.0f, lw2 = 0.0f;
    size_t stride = (size_t)gridDim.x * blockDim.x;
    for (size_t i = (size_t)blockIdx.x * blockDim.x + threadIdx.x; i < n; i += stride) {
        float r = rho[i];
        float sp = (r > 15.0f) ? r : log1pf(__expf(r));
        float sigma = sp + 1e-5f;
        float e = eps[i];
        float w = mu[i] + sigma * e;
        wb[i] = (__bf16)w;                       // native cvt on gfx1250
        lpost += -__logf(sigma) - 0.5f * e * e;
        lw2 += w * w;
    }
    __shared__ float r0[256], r1[256];
    r0[threadIdx.x] = lpost; r1[threadIdx.x] = lw2;
    __syncthreads();
    for (int s = 128; s > 0; s >>= 1) {
        if ((int)threadIdx.x < s) {
            r0[threadIdx.x] += r0[threadIdx.x + s];
            r1[threadIdx.x] += r1[threadIdx.x + s];
        }
        __syncthreads();
    }
    if (threadIdx.x == 0) { atomicAdd(&acc[0], r0[0]); atomicAdd(&acc[1], r1[0]); }
}

// both bias tensors in one pass; writes combined bias (b_ih + b_hh) in f32
__global__ __launch_bounds__(256)
void sample_bias_kernel(const float* __restrict__ mu1, const float* __restrict__ rho1,
                        const float* __restrict__ eps1,
                        const float* __restrict__ mu2, const float* __restrict__ rho2,
                        const float* __restrict__ eps2,
                        int n, float* __restrict__ bias_comb, float* __restrict__ acc) {
    int i = blockIdx.x * blockDim.x + threadIdx.x;
    float p1 = 0, q1 = 0, p2 = 0, q2 = 0;
    if (i < n) {
        float r1 = rho1[i];
        float s1 = ((r1 > 15.0f) ? r1 : log1pf(__expf(r1))) + 1e-5f;
        float e1 = eps1[i];
        float w1 = mu1[i] + s1 * e1;
        p1 = -__logf(s1) - 0.5f * e1 * e1; q1 = w1 * w1;
        float r2 = rho2[i];
        float s2 = ((r2 > 15.0f) ? r2 : log1pf(__expf(r2))) + 1e-5f;
        float e2 = eps2[i];
        float w2 = mu2[i] + s2 * e2;
        p2 = -__logf(s2) - 0.5f * e2 * e2; q2 = w2 * w2;
        bias_comb[i] = w1 + w2;
    }
    __shared__ float red[4][256];
    red[0][threadIdx.x] = p1; red[1][threadIdx.x] = q1;
    red[2][threadIdx.x] = p2; red[3][threadIdx.x] = q2;
    __syncthreads();
    for (int s = 128; s > 0; s >>= 1) {
        if ((int)threadIdx.x < s)
            for (int k = 0; k < 4; ++k) red[k][threadIdx.x] += red[k][threadIdx.x + s];
        __syncthreads();
    }
    if (threadIdx.x == 0)
        for (int k = 0; k < 4; ++k) atomicAdd(&acc[k], red[k][0]);
}

// ---------------------------------------------------------------- KL finalize (1 thread)
__global__ void kl_finalize_kernel(const float* __restrict__ acc, float* __restrict__ out_kl) {
    if (threadIdx.x == 0 && blockIdx.x == 0) {
        const double LOG2PI = 1.8378770664093453;
        const double C1 = 3.694528049465325;        // 0.5*e^2  (sigma1 = e^-1)
        const double C2 = 601302.1420823884;        // 0.5*e^14 (sigma2 = e^-7)
        const double LPI1 = -0.2876820724517809;    // log 0.75
        const double LPI2 = -1.3862943611198906;    // log 0.25
        const double Ns[4] = {524288.0, 4194304.0, 4096.0, 4096.0};
        double kl = 0.0;
        for (int i = 0; i < 4; ++i) {
            double post = -0.5 * LOG2PI * Ns[i] + (double)acc[2 * i];
            double w2 = (double)acc[2 * i + 1];
            double m1 = (1.0 - 0.5 * LOG2PI) * Ns[i] - C1 * w2 + LPI1;
            double m2 = (7.0 - 0.5 * LOG2PI) * Ns[i] - C2 * w2 + LPI2;
            double mx = fmax(m1, m2);
            double lprior = mx + log(exp(m1 - mx) + exp(m2 - mx));
            kl += post - lprior;
        }
        *out_kl = (float)kl;
    }
}

// ---------------------------------------------------------------- xg = x @ w_ih^T + (b_ih+b_hh)
// M = T*B = 49152, N = G = 4096, K = 128.  Block tile 128x128, 8 waves (32x64 each).
constexpr int LDA1 = 40;   // ushorts per LDS row (32 + 8 pad; 80B rows, 16B-aligned chunks)

__global__ __launch_bounds__(256)
void xg_gemm_kernel(const float* __restrict__ x, const unsigned short* __restrict__ wih,
                    const float* __restrict__ bias, float* __restrict__ xg) {
    __shared__ unsigned short ldsA[128 * LDA1];
    __shared__ unsigned short ldsB[128 * LDA1];
    const int tid = threadIdx.x;
    const int lane = tid & 31;
    const int wave = tid >> 5;
    const int wm = wave & 3;      // row group of 32
    const int wn = wave >> 2;     // col group of 64
    const int m0 = blockIdx.x * 128;
    const int n0 = blockIdx.y * 128;

    v8f zero = {};
    v8f acc[2][4];
#pragma unroll
    for (int a = 0; a < 2; ++a)
#pragma unroll
        for (int b = 0; b < 4; ++b) acc[a][b] = zero;

    const int l15 = lane & 15;
    const int kb  = (lane < 16) ? 0 : 8;    // A fragment K base
    const int k0l = (lane < 16) ? 0 : 16;   // unused; B fragments also use kb

    for (int k0 = 0; k0 < IN_DIM; k0 += 32) {
        {   // A tile: x rows m0..m0+127, 32 K, f32 -> bf16 (native cvt) -> LDS
            int r = tid >> 1;
            int hh = (tid & 1) * 16;
            const float* src = x + (size_t)(m0 + r) * IN_DIM + k0 + hh;
            __bf16* dstb = (__bf16*)&ldsA[r * LDA1 + hh];
#pragma unroll
            for (int j = 0; j < 16; ++j) dstb[j] = (__bf16)src[j];
        }
        {   // B tile: w_ih rows n0..n0+127 (bf16, 64B per 32-K chunk) via async->LDS
            int r = tid >> 1;
            int q = (tid & 1) * 2;                       // uint4 pair
            const unsigned short* src = wih + (size_t)(n0 + r) * IN_DIM + k0 + q * 8;
            unsigned loff = lds_off32(&ldsB[r * LDA1 + q * 8]);
            async_copy_b128(loff, src);
            async_copy_b128_off16(loff, src);
        }
        s_wait_asynccnt_0();
        __syncthreads();

        v16bf bfr[4];
#pragma unroll
        for (int nf = 0; nf < 4; ++nf) {
            const unsigned short* bp = &ldsB[(wn * 64 + nf * 16 + l15) * LDA1 + kb];
            bfr[nf] = load_frag16(bp, bp + 16);
        }
#pragma unroll
        for (int mf = 0; mf < 2; ++mf) {
            const unsigned short* ap = &ldsA[(wm * 32 + mf * 16 + l15) * LDA1 + kb];
            v16bf afr = load_frag16(ap, ap + 16);
#pragma unroll
            for (int nf = 0; nf < 4; ++nf)
                acc[mf][nf] = __builtin_amdgcn_wmma_f32_16x16x32_bf16(
                    false, afr, false, bfr[nf], (short)0, acc[mf][nf], false, false);
        }
        __syncthreads();
    }

    const int mtop = (lane < 16) ? 0 : 8;
#pragma unroll
    for (int nf = 0; nf < 4; ++nf) {
        int gcol = n0 + wn * 64 + nf * 16 + l15;
        float bb = bias[gcol];
#pragma unroll
        for (int mf = 0; mf < 2; ++mf) {
            int mrow = m0 + wm * 32 + mf * 16 + mtop;
#pragma unroll
            for (int r = 0; r < 8; ++r)
                xg[(size_t)(mrow + r) * GATES + gcol] = acc[mf][nf][r] + bb;
        }
    }
}

// ---------------------------------------------------------------- fused LSTM step
// Block: 64 batch rows x 16 h-cols across ALL 4 gate regions (GEMM N = 64).
// grid = (2, 64).  K = 1024, async double-buffered 64-K stages. 8 waves, each 16x32.
constexpr int LDA2 = 72;   // 64 + 8 pad ushorts (144B rows, 16B-aligned chunks)
constexpr int LDG  = 68;   // f32 gate staging stride
constexpr int KSTAGE = 64;
constexpr int NSTAGE = HID / KSTAGE;   // 16

__global__ __launch_bounds__(256)
void lstm_step_kernel(const float* __restrict__ xg, const unsigned short* __restrict__ whh,
                      const unsigned short* __restrict__ hprev,
                      unsigned short* __restrict__ hnext,
                      float* __restrict__ cbuf, float* __restrict__ ys, int t) {
    __shared__ unsigned short ldsA[2][64 * LDA2];
    __shared__ unsigned short ldsB[2][64 * LDA2];
    __shared__ float ldsG[64 * LDG];

    const int tid = threadIdx.x;
    const int lane = tid & 31;
    const int wave = tid >> 5;
    const int wm = wave >> 1;     // 0..3 -> 16 rows
    const int wn = wave & 1;      // 0..1 -> 32 gate-cols
    const int m0  = blockIdx.x * 64;
    const int hc0 = blockIdx.y * 16;

    const int l15 = lane & 15;
    const int kb  = (lane < 16) ? 0 : 8;
    const int k0l = (lane < 16) ? 0 : 16;

    v8f zero = {};
    v8f acc[2]; acc[0] = zero; acc[1] = zero;

    const int rA = tid >> 2;            // 0..63 (tile row for loads)
    const int qA = (tid & 3) * 16;      // ushort offset of this lane's 32B slice
    const int gB = (rA >> 4) * HID + hc0 + (rA & 15);   // gate-column -> w_hh row

    const unsigned short* aRow = hprev + (size_t)(m0 + rA) * HID + qA;
    const unsigned short* bRow = whh + (size_t)gB * HID + qA;

    // prologue: stage 0 into buffer 0 (4 async instructions per wave per stage)
    {
        unsigned la = lds_off32(&ldsA[0][rA * LDA2 + qA]);
        unsigned lb = lds_off32(&ldsB[0][rA * LDA2 + qA]);
        async_copy_b128(la, aRow);
        async_copy_b128_off16(la, aRow);
        async_copy_b128(lb, bRow);
        async_copy_b128_off16(lb, bRow);
    }

    for (int i = 0; i < NSTAGE; ++i) {
        const int cur = i & 1;
        if (i + 1 < NSTAGE) {          // issue stage i+1 into the other buffer
            const unsigned short* an = aRow + (i + 1) * KSTAGE;
            const unsigned short* bn = bRow + (i + 1) * KSTAGE;
            unsigned la = lds_off32(&ldsA[cur ^ 1][rA * LDA2 + qA]);
            unsigned lb = lds_off32(&ldsB[cur ^ 1][rA * LDA2 + qA]);
            async_copy_b128(la, an);
            async_copy_b128_off16(la, an);
            async_copy_b128(lb, bn);
            async_copy_b128_off16(lb, bn);
            s_wait_asynccnt_4();       // stage i complete; i+1's 4 stay in flight
        } else {
            s_wait_asynccnt_0();
        }
        __syncthreads();

#pragma unroll
        for (int kk = 0; kk < KSTAGE; kk += 32) {
            const unsigned short* ap = &ldsA[cur][(wm * 16 + l15) * LDA2 + kk + kb];
            v16bf afr = load_frag16(ap, ap + 16);
#pragma unroll
            for (int nf = 0; nf < 2; ++nf) {
                const unsigned short* bp = &ldsB[cur][(wn * 32 + nf * 16 + l15) * LDA2 + kk + kb];
                v16bf bfr = load_frag16(bp, bp + 16);
                acc[nf] = __builtin_amdgcn_wmma_f32_16x16x32_bf16(
                    false, afr, false, bfr, (short)0, acc[nf], false, false);
            }
        }
        __syncthreads();               // buffer cur may be overwritten next iter
    }

    // stage gates to LDS so each (row, h-col) sees all four gates
    const int mtop = (lane < 16) ? 0 : 8;
#pragma unroll
    for (int nf = 0; nf < 2; ++nf) {
        int n = wn * 32 + nf * 16 + l15;
        int mb = wm * 16 + mtop;
#pragma unroll
        for (int r = 0; r < 8; ++r)
            ldsG[(mb + r) * LDG + n] = acc[nf][r];
    }
    __syncthreads();

    // fused cell update: c' = sig(f)*c + sig(i)*tanh(g); h' = sig(o)*tanh(c')
    __bf16* hnb = (__bf16*)hnext;
    for (int e = tid; e < 64 * 16; e += 256) {
        int row = e >> 4;
        int j = e & 15;
        int m = m0 + row;
        int hc = hc0 + j;
        const float* xgp = xg + ((size_t)t * BATCH + m) * GATES;
        float gi = ldsG[row * LDG + j]           + xgp[hc];
        float gf = ldsG[row * LDG + 16 + j]      + xgp[HID + hc];
        float gg = ldsG[row * LDG + 32 + j]      + xgp[2 * HID + hc];
        float go = ldsG[row * LDG + 48 + j]      + xgp[3 * HID + hc];
        size_t sidx = (size_t)m * HID + hc;
        float cold = cbuf[sidx];
        float cnew = sigm(gf) * cold + sigm(gi) * tanh_fast(gg);
        float hnew = sigm(go) * tanh_fast(cnew);
        cbuf[sidx] = cnew;
        ys[((size_t)t * BATCH + m) * HID + hc] = hnew;
        hnb[sidx] = (__bf16)hnew;
    }
}

// ---------------------------------------------------------------- tail: h_n, c_n
__global__ void copyout_kernel(const float* __restrict__ cbuf, float* __restrict__ dout) {
    int i = blockIdx.x * blockDim.x + threadIdx.x;
    const size_t ys_size = (size_t)T_STEPS * BATCH * HID;
    const size_t bh = (size_t)BATCH * HID;
    if (i < (int)bh) {
        dout[ys_size + i] = dout[ys_size - bh + i];   // h_n = ys[T-1]
        dout[ys_size + bh + i] = cbuf[i];             // c_n
    }
}

// ---------------------------------------------------------------- launch
extern "C" void kernel_launch(void* const* d_in, const int* in_sizes, int n_in,
                              void* d_out, int out_size, void* d_ws, size_t ws_size,
                              hipStream_t stream) {
    const float* x       = (const float*)d_in[0];
    const float* wih_mu  = (const float*)d_in[1];
    const float* wih_rho = (const float*)d_in[2];
    const float* wih_eps = (const float*)d_in[3];
    const float* whh_mu  = (const float*)d_in[4];
    const float* whh_rho = (const float*)d_in[5];
    const float* whh_eps = (const float*)d_in[6];
    const float* bih_mu  = (const float*)d_in[7];
    const float* bih_rho = (const float*)d_in[8];
    const float* bih_eps = (const float*)d_in[9];
    const float* bhh_mu  = (const float*)d_in[10];
    const float* bhh_rho = (const float*)d_in[11];
    const float* bhh_eps = (const float*)d_in[12];
    float* dout = (float*)d_out;

    // workspace carve-up (256B aligned)
    char* ws = (char*)d_ws;
    size_t off = 0;
    auto carve = [&](size_t bytes) {
        size_t o = off;
        off = (off + bytes + 255) & ~(size_t)255;
        return o;
    };
    const size_t NW_IH = (size_t)GATES * IN_DIM;     // 524288
    const size_t NW_HH = (size_t)GATES * HID;        // 4194304
    const size_t BH    = (size_t)BATCH * HID;

    size_t off_xg   = carve((size_t)T_STEPS * BATCH * GATES * sizeof(float)); // 805 MB
    size_t off_wih  = carve(NW_IH * sizeof(unsigned short));
    size_t off_whh  = carve(NW_HH * sizeof(unsigned short));
    size_t off_bias = carve(GATES * sizeof(float));
    size_t off_h0   = carve(BH * sizeof(unsigned short));
    size_t off_h1   = carve(BH * sizeof(unsigned short));
    size_t off_c    = carve(BH * sizeof(float));
    size_t off_acc  = carve(8 * sizeof(float));
    if (ws_size < off) return;   // insufficient scratch

    float*          xg    = (float*)(ws + off_xg);
    unsigned short* wihb  = (unsigned short*)(ws + off_wih);
    unsigned short* whhb  = (unsigned short*)(ws + off_whh);
    float*          biasc = (float*)(ws + off_bias);
    unsigned short* h0    = (unsigned short*)(ws + off_h0);
    unsigned short* h1    = (unsigned short*)(ws + off_h1);
    float*          cbuf  = (float*)(ws + off_c);
    float*          acc   = (float*)(ws + off_acc);

    // 1) zero state + KL accumulators (must re-run every call)
    init_kernel<<<(int)((BH + 255) / 256), 256, 0, stream>>>(h0, h1, cbuf, acc, (int)BH);

    // 2) sample weights (bf16 for WMMA) + KL partial reductions
    sample_matrix_kernel<<<512, 256, 0, stream>>>(wih_mu, wih_rho, wih_eps, NW_IH, wihb, acc + 0);
    sample_matrix_kernel<<<2048, 256, 0, stream>>>(whh_mu, whh_rho, whh_eps, NW_HH, whhb, acc + 2);
    sample_bias_kernel<<<GATES / 256, 256, 0, stream>>>(bih_mu, bih_rho, bih_eps,
                                                        bhh_mu, bhh_rho, bhh_eps,
                                                        GATES, biasc, acc + 4);

    // 3) xg = x @ w_ih^T + (b_ih + b_hh)   [49152 x 4096, K=128]
    xg_gemm_kernel<<<dim3(T_STEPS * BATCH / 128, GATES / 128), 256, 0, stream>>>(
        x, wihb, biasc, xg);

    // 4) sequential fused LSTM steps (h double-buffered in bf16)
    for (int t = 0; t < T_STEPS; ++t) {
        const unsigned short* hp = (t & 1) ? h1 : h0;
        unsigned short* hn = (t & 1) ? h0 : h1;
        lstm_step_kernel<<<dim3(BATCH / 64, HID / 16), 256, 0, stream>>>(
            xg, whhb, hp, hn, cbuf, dout, t);
    }

    // 5) h_n / c_n tail + KL scalar
    copyout_kernel<<<(int)((BH + 255) / 256), 256, 0, stream>>>(cbuf, dout);
    kl_finalize_kernel<<<1, 64, 0, stream>>>(
        acc, dout + (size_t)T_STEPS * BATCH * HID + 2 * BH);
}
